// LinOSSBlock_35862976921827
// MI455X (gfx1250) — matrix operand order repaired and checked by reading, compiled, not run on hardware
//
#include <hip/hip_runtime.h>

#define B_SZ   8
#define LSEQ   4096
#define HDIM   128
#define PDIM   256
#define TCHUNK 128
#define NCHUNK (LSEQ / TCHUNK)     // 32
#define BLROWS (B_SZ * LSEQ)       // 32768

typedef __attribute__((ext_vector_type(16))) _Float16 v16h;
typedef __attribute__((ext_vector_type(8)))  float    v8f;

// D = A(16x32,f16) * B(32x16,f16) + C(16x16,f32), wave32 WMMA (CDNA5)
__device__ __forceinline__ v8f wmma16(v16h a, v16h b, v8f c) {
  return __builtin_amdgcn_wmma_f32_16x16x32_f16(
      /*neg_a=*/false, a, /*neg_b=*/false, b,
      /*c_mod=*/(short)0, c, /*reuse_a=*/false, /*reuse_b=*/false);
}

// A fragment (16x32 f16) from a row-major f32 matrix, per ISA 05_wmma layout:
// lane m<16: row m, K = {0..7, 16..23}; lane m+16: row m, K = {8..15, 24..31}
__device__ __forceinline__ v16h load_a_f32(const float* __restrict__ base, int stride,
                                           int r0, int k0, int lane) {
  const int m   = lane & 15;
  const int hiK = (lane >> 4) << 3;                 // 0 or 8
  const float* rowp = base + (size_t)(r0 + m) * stride + (k0 + hiK);
  v16h a;
#pragma unroll
  for (int e = 0; e < 16; ++e) {
    const int kk = e + (e & 8);                     // 0..7, 16..23
    a[e] = (_Float16)rowp[kk];
  }
  return a;
}

// B fragment (32x16 f16). Operand stored as plane[n*stride + k] (i.e. B_op[k][n]).
// lane l<16: col l, K = {0..7, 16..23}; lane l>=16: col l-16, K = {8..15, 24..31}
__device__ __forceinline__ v16h load_b_f16(const _Float16* __restrict__ plane, int stride,
                                           int n0, int k0, int lane) {
  const int n   = lane & 15;
  const int hiK = (lane >> 4) << 3;                 // 0 or 8
  const _Float16* colp = plane + (size_t)(n0 + n) * stride + (k0 + hiK);
  v16h b;
#pragma unroll
  for (int e = 0; e < 16; ++e) b[e] = colp[e + (e & 8)];
  return b;
}

// ---------------------------------------------------------------------------
// Phase 0: per-p parameter projection, M^t power table, operand conversion.
// Folds dts into B (F1 = dts * x@B^T) and the minus sign into Cim (f16 WMMA
// has no A/B NEG modifier on CDNA5; only C-neg is legal).
// ---------------------------------------------------------------------------
__global__ void prep_kernel(const float* __restrict__ A_diag, const float* __restrict__ G_diag,
                            const float* __restrict__ dt,
                            const float* __restrict__ Bsrc, const float* __restrict__ Csrc,
                            _Float16* __restrict__ Bre_h, _Float16* __restrict__ Bim_h,
                            _Float16* __restrict__ Cre_h, _Float16* __restrict__ Cimn_h,
                            float* __restrict__ par, float2* __restrict__ Mpow2,
                            float* __restrict__ MTfull) {
  __shared__ float sdts[PDIM];
  const int p = threadIdx.x;
  {
    float dts  = 1.0f / (1.0f + __expf(-dt[p]));
    float Glow = dts * A_diag[p];
    float G    = Glow + fmaxf(G_diag[p] - Glow, 0.0f);
    float Alow = 0.25f * G * G;
    float Av   = Alow + fmaxf(A_diag[p] - Alow, 0.0f);
    float m11  = 1.0f - dts * G;
    float m12  = -dts * Av;
    float m21  = dts;                                // m22 == 1
    par[p] = m11; par[PDIM + p] = m12; par[2 * PDIM + p] = m21;
    sdts[p] = dts;
    // M^t, t = 1..TCHUNK; store row2 (needed by fixup) and full M^T (carry chain)
    float a = m11, b = m12, c = m21, d = 1.0f;
    for (int t = 0; t < TCHUNK; ++t) {
      Mpow2[t * PDIM + p] = make_float2(c, d);       // row2 of M^{t+1}
      if (t == TCHUNK - 1) {
        MTfull[4*p] = a; MTfull[4*p+1] = b; MTfull[4*p+2] = c; MTfull[4*p+3] = d;
      }
      float na = m11*a + m12*c, nb = m11*b + m12*d;
      float nc = m21*a + c,     nd = m21*b + d;
      a = na; b = nb; c = nc; d = nd;
    }
  }
  __syncthreads();
  for (int i = threadIdx.x; i < PDIM * HDIM; i += blockDim.x) {  // B: (P,H,2)
    float s = sdts[i / HDIM];
    Bre_h[i] = (_Float16)(s * Bsrc[2*i]);
    Bim_h[i] = (_Float16)(s * Bsrc[2*i + 1]);
  }
  for (int i = threadIdx.x; i < HDIM * PDIM; i += blockDim.x) {  // C: (H,P,2)
    Cre_h[i]  = (_Float16)( Csrc[2*i]);
    Cimn_h[i] = (_Float16)(-Csrc[2*i + 1]);
  }
}

// ---------------------------------------------------------------------------
// Phase 1: F1 = dts * (x @ B^T), complex (re/im planes). One 16x16 tile/wave.
// ---------------------------------------------------------------------------
__global__ void gemm1_kernel(const float* __restrict__ x,
                             const _Float16* __restrict__ Bre_h,
                             const _Float16* __restrict__ Bim_h,
                             float* __restrict__ F1re, float* __restrict__ F1im) {
  const int lane = threadIdx.x & 31;
  const int wv   = blockIdx.x * (blockDim.x >> 5) + (threadIdx.x >> 5);
  const int r0   = (wv >> 4) << 4;                  // bl-tile
  const int c0   = (wv & 15) << 4;                  // p-tile
  v8f accr = {}; v8f acci = {};
#pragma unroll
  for (int k0 = 0; k0 < HDIM; k0 += 32) {
    v16h a  = load_a_f32(x,     HDIM, r0, k0, lane);
    v16h br = load_b_f16(Bre_h, HDIM, c0, k0, lane);
    v16h bi = load_b_f16(Bim_h, HDIM, c0, k0, lane);
    accr = wmma16(a, br, accr);
    acci = wmma16(a, bi, acci);
  }
  const int col = c0 + (lane & 15);
  const int rb  = r0 + ((lane >> 4) << 3);
#pragma unroll
  for (int r = 0; r < 8; ++r) {
    size_t o = (size_t)(rb + r) * PDIM + col;
    F1re[o] = accr[r];
    F1im[o] = acci[r];
  }
}

// ---------------------------------------------------------------------------
// Phase 2a: per-(b,p,chunk) local scan, zero carry-in; overwrite F1 in place
// with the local x-state; save full end-of-chunk state for the carry chain.
// ---------------------------------------------------------------------------
__global__ void scan_local_kernel(float* __restrict__ F1re, float* __restrict__ F1im,
                                  const float* __restrict__ par, float* __restrict__ le) {
  const int p = threadIdx.x;
  const int b = blockIdx.x / NCHUNK;
  const int c = blockIdx.x % NCHUNK;
  const float m11 = par[p], m12 = par[PDIM + p], m21 = par[2 * PDIM + p];
  float zr = 0.f, xr = 0.f, zi = 0.f, xi = 0.f;
  size_t idx = ((size_t)b * LSEQ + (size_t)c * TCHUNK) * PDIM + p;
  for (int t = 0; t < TCHUNK; ++t, idx += PDIM) {
    float fr = F1re[idx], fi = F1im[idx];
    float nzr = fmaf(m11, zr, fmaf(m12, xr, fr));
    float nxr = fmaf(m21, zr, xr);
    float nzi = fmaf(m11, zi, fmaf(m12, xi, fi));
    float nxi = fmaf(m21, zi, xi);
    zr = nzr; xr = nxr; zi = nzi; xi = nxi;
    F1re[idx] = xr;
    F1im[idx] = xi;
  }
  const int BCP = B_SZ * NCHUNK * PDIM;
  const int o   = (b * NCHUNK + c) * PDIM + p;
  le[o] = zr; le[BCP + o] = xr; le[2*BCP + o] = zi; le[3*BCP + o] = xi;
}

// Phase 2b: sequential carry propagation across the 32 chunks (per b,p):
// carryIn(0)=0; carryIn(c+1) = M^T * carryIn(c) + localEnd(c)
__global__ void carry_kernel(const float* __restrict__ le, const float* __restrict__ MTfull,
                             float* __restrict__ ca) {
  const int p = threadIdx.x;
  const int b = blockIdx.x;
  const float a  = MTfull[4*p],   bb = MTfull[4*p+1];
  const float cc = MTfull[4*p+2], dd = MTfull[4*p+3];
  const int BCP = B_SZ * NCHUNK * PDIM;
  float zr = 0.f, xr = 0.f, zi = 0.f, xi = 0.f;
  for (int c = 0; c < NCHUNK; ++c) {
    const int o = (b * NCHUNK + c) * PDIM + p;
    ca[o] = zr; ca[BCP + o] = xr; ca[2*BCP + o] = zi; ca[3*BCP + o] = xi;
    float lzr = le[o], lxr = le[BCP + o], lzi = le[2*BCP + o], lxi = le[3*BCP + o];
    float nzr = fmaf(a,  zr, fmaf(bb, xr, lzr));
    float nxr = fmaf(cc, zr, fmaf(dd, xr, lxr));
    float nzi = fmaf(a,  zi, fmaf(bb, xi, lzi));
    float nxi = fmaf(cc, zi, fmaf(dd, xi, lxi));
    zr = nzr; xr = nxr; zi = nzi; xi = nxi;
  }
}

// Phase 2c: fixup. x_global[c0+t] = x_local + row2(M^{t+1}) . carryIn(chunk)
__global__ void fixup_kernel(float* __restrict__ F1re, float* __restrict__ F1im,
                             const float2* __restrict__ Mpow2, const float* __restrict__ ca) {
  const int p = threadIdx.x;
  const int b = blockIdx.x / NCHUNK;
  const int c = blockIdx.x % NCHUNK;
  if (c == 0) return;                                // carry-in is zero
  const int BCP = B_SZ * NCHUNK * PDIM;
  const int o   = (b * NCHUNK + c) * PDIM + p;
  const float czr = ca[o], cxr = ca[BCP + o], czi = ca[2*BCP + o], cxi = ca[3*BCP + o];
  size_t idx = ((size_t)b * LSEQ + (size_t)c * TCHUNK) * PDIM + p;
  for (int t = 0; t < TCHUNK; ++t, idx += PDIM) {
    float2 m = Mpow2[t * PDIM + p];
    F1re[idx] = fmaf(m.x, czr, fmaf(m.y, cxr, F1re[idx]));
    F1im[idx] = fmaf(m.x, czi, fmaf(m.y, cxi, F1im[idx]));
  }
}

// ---------------------------------------------------------------------------
// Phase 3: out = ys_re @ Cre^T + ys_im @ (-Cim)^T + D*x
// ---------------------------------------------------------------------------
__global__ void gemm2_kernel(const float* __restrict__ ysre, const float* __restrict__ ysim,
                             const _Float16* __restrict__ Cre_h, const _Float16* __restrict__ Cimn_h,
                             const float* __restrict__ x, const float* __restrict__ Dv,
                             float* __restrict__ out) {
  const int lane = threadIdx.x & 31;
  const int wv   = blockIdx.x * (blockDim.x >> 5) + (threadIdx.x >> 5);
  const int r0   = (wv >> 3) << 4;                   // bl-tile
  const int c0   = (wv & 7) << 4;                    // h-tile
  v8f acc = {};
#pragma unroll
  for (int k0 = 0; k0 < PDIM; k0 += 32) {
    v16h ar = load_a_f32(ysre,   PDIM, r0, k0, lane);
    v16h ai = load_a_f32(ysim,   PDIM, r0, k0, lane);
    v16h br = load_b_f16(Cre_h,  PDIM, c0, k0, lane);
    v16h bi = load_b_f16(Cimn_h, PDIM, c0, k0, lane);
    acc = wmma16(ar, br, acc);
    acc = wmma16(ai, bi, acc);
  }
  const int col = c0 + (lane & 15);
  const int rb  = r0 + ((lane >> 4) << 3);
  const float dscale = Dv[col];
#pragma unroll
  for (int r = 0; r < 8; ++r) {
    size_t o = (size_t)(rb + r) * HDIM + col;
    out[o] = acc[r] + dscale * x[o];
  }
}

extern "C" void kernel_launch(void* const* d_in, const int* in_sizes, int n_in,
                              void* d_out, int out_size, void* d_ws, size_t ws_size,
                              hipStream_t stream) {
  (void)in_sizes; (void)n_in; (void)out_size; (void)ws_size;
  const float* x      = (const float*)d_in[0];
  const float* A_diag = (const float*)d_in[1];
  const float* G_diag = (const float*)d_in[2];
  const float* dt     = (const float*)d_in[3];
  const float* Bsrc   = (const float*)d_in[4];
  const float* Csrc   = (const float*)d_in[5];
  const float* Dv     = (const float*)d_in[6];
  float* out = (float*)d_out;

  const size_t S = (size_t)BLROWS * PDIM;            // 8,388,608 floats per plane
  char* w = (char*)d_ws;
  float*    F1re   = (float*)w;     w += S * sizeof(float);        // 32 MB
  float*    F1im   = (float*)w;     w += S * sizeof(float);        // 32 MB
  _Float16* Bre_h  = (_Float16*)w;  w += (size_t)PDIM * HDIM * 2;  // 64 KB
  _Float16* Bim_h  = (_Float16*)w;  w += (size_t)PDIM * HDIM * 2;
  _Float16* Cre_h  = (_Float16*)w;  w += (size_t)HDIM * PDIM * 2;
  _Float16* Cimn_h = (_Float16*)w;  w += (size_t)HDIM * PDIM * 2;
  float*    par    = (float*)w;     w += 3 * PDIM * sizeof(float);
  float2*   Mpow2  = (float2*)w;    w += (size_t)TCHUNK * PDIM * sizeof(float2);
  float*    MTfull = (float*)w;     w += 4 * PDIM * sizeof(float);
  float*    le     = (float*)w;     w += 4 * (size_t)B_SZ * NCHUNK * PDIM * sizeof(float);
  float*    ca     = (float*)w;     w += 4 * (size_t)B_SZ * NCHUNK * PDIM * sizeof(float);

  prep_kernel<<<1, PDIM, 0, stream>>>(A_diag, G_diag, dt, Bsrc, Csrc,
                                      Bre_h, Bim_h, Cre_h, Cimn_h, par, Mpow2, MTfull);
  // 2048 x 16 tiles, 4 waves/block
  gemm1_kernel<<<(BLROWS / 16) * (PDIM / 16) / 4, 128, 0, stream>>>(x, Bre_h, Bim_h, F1re, F1im);
  scan_local_kernel<<<B_SZ * NCHUNK, PDIM, 0, stream>>>(F1re, F1im, par, le);
  carry_kernel<<<B_SZ, PDIM, 0, stream>>>(le, MTfull, ca);
  fixup_kernel<<<B_SZ * NCHUNK, PDIM, 0, stream>>>(F1re, F1im, Mpow2, ca);
  // 2048 x 8 tiles, 4 waves/block
  gemm2_kernel<<<(BLROWS / 16) * (HDIM / 16) / 4, 128, 0, stream>>>(F1re, F1im, Cre_h, Cimn_h,
                                                                    x, Dv, out);
}
